// SAModule_51762945851724
// MI455X (gfx1250) — matrix operand order repaired and checked by reading, compiled, not run on hardware
//
#include <hip/hip_runtime.h>
#include <hip/hip_bf16.h>

typedef __attribute__((ext_vector_type(16))) _Float16 v16h;
typedef __attribute__((ext_vector_type(8)))  float    v8f;

#define NPTS 8192
#define MSMP 4096
#define KNBR 64
#define R2   0.04f

// ---------------------------------------------------------------------------
// WMMA helpers: ISA 7.12.2 layouts.
// A (16x32 f16): lane<16 -> row M=lane, K={0..7,16..23}; lane>=16 same row,
// K={8..15,24..31}. VGPR pair j covers K=(j>>2)*16 + hi*8 + (j&3)*2.
// B (32x16 f16): N col = lane&15, VGPR pair j covers the same K as in A.
// C/D (16x16 f32): lane<16 -> N=lane, VGPR j -> M=j; lane>=16 -> M=8+j.
// ---------------------------------------------------------------------------
__device__ __forceinline__ v16h load_A_f16(const _Float16* base, int ld) {
  const int lane = threadIdx.x & 31;
  const int row  = lane & 15;
  const int hi   = lane >> 4;
  v16h a;
#pragma unroll
  for (int j = 0; j < 8; ++j) {
    const int k = ((j >> 2) * 16) + hi * 8 + (j & 3) * 2;
    a[2 * j]     = base[row * ld + k];
    a[2 * j + 1] = base[row * ld + k + 1];
  }
  return a;
}

__device__ __forceinline__ v16h load_B_f16(const _Float16* base, int ld, int n0) {
  const int lane = threadIdx.x & 31;  // N col = lane&15, K as in A
  const int col  = lane & 15;
  const int hi   = lane >> 4;
  v16h b;
#pragma unroll
  for (int j = 0; j < 8; ++j) {
    const int k = ((j >> 2) * 16) + hi * 8 + (j & 3) * 2;
    b[2 * j]     = base[k * ld + n0 + col];
    b[2 * j + 1] = base[(k + 1) * ld + n0 + col];
  }
  return b;
}

__device__ __forceinline__ v8f wmma32(v16h a, v16h b, v8f c) {
  return __builtin_amdgcn_wmma_f32_16x16x32_f16(false, a, false, b, (short)0, c,
                                                false, false);
}

// Async copy of one 16B chunk: global (saddr + voffset) -> LDS (vdst offset).
// Tracked by ASYNCcnt; fenced below with s_wait_asynccnt 0.
__device__ __forceinline__ void async_copy16(unsigned lds_off, int goff,
                                             const void* gbase) {
  asm volatile("global_load_async_to_lds_b128 %0, %1, %2"
               :
               : "v"(lds_off), "v"(goff), "s"(gbase)
               : "memory");
}

// ---------------------------------------------------------------------------
// Kernel 1: farthest point sampling. Single persistent workgroup, pos + mind
// resident in LDS (128 KB; CDNA5 allows up to 320 KB per workgroup).
// ---------------------------------------------------------------------------
__global__ __launch_bounds__(1024) void fps_kernel(const float* __restrict__ pos,
                                                   int* __restrict__ idx_out) {
  __shared__ float sx[NPTS], sy[NPTS], sz[NPTS];
  __shared__ float mind[NPTS];
  __shared__ unsigned long long waveBest[32];
  __shared__ int sLast;

  const int t = threadIdx.x;
  const int lane = t & 31;
  const int wid = t >> 5;

  for (int i = t; i < NPTS; i += 1024) {
    sx[i] = pos[i * 3 + 0];
    sy[i] = pos[i * 3 + 1];
    sz[i] = pos[i * 3 + 2];
    mind[i] = 1e30f;
  }
  if (t == 0) idx_out[0] = 0;
  __syncthreads();

  int last = 0;
  for (int s = 1; s < MSMP; ++s) {
    const float lx = sx[last], ly = sy[last], lz = sz[last];
    unsigned long long best = 0ull;
    for (int i = t; i < NPTS; i += 1024) {   // 8 points per thread, owner-only
      const float dx = sx[i] - lx, dy = sy[i] - ly, dz = sz[i] - lz;
      float m = fminf(mind[i], dx * dx + dy * dy + dz * dz);
      mind[i] = m;
      // mind >= 0 -> float bits monotonic; ~i breaks ties toward lowest index
      unsigned long long key =
          ((unsigned long long)__float_as_uint(m) << 32) |
          (unsigned)(0xFFFFFFFFu - (unsigned)i);
      best = (key > best) ? key : best;
    }
#pragma unroll
    for (int off = 16; off; off >>= 1) {
      unsigned long long o = __shfl_xor(best, off, 32);
      best = (o > best) ? o : best;
    }
    if (lane == 0) waveBest[wid] = best;
    __syncthreads();
    if (t < 32) {
      unsigned long long b = waveBest[t];
#pragma unroll
      for (int off = 16; off; off >>= 1) {
        unsigned long long o = __shfl_xor(b, off, 32);
        b = (o > b) ? o : b;
      }
      if (t == 0) {
        const int nxt = (int)(0xFFFFFFFFu - (unsigned)b);
        idx_out[s] = nxt;
        sLast = nxt;
      }
    }
    __syncthreads();
    last = sLast;
  }
}

// ---------------------------------------------------------------------------
// Kernel 2: top-64 nearest within radius, one wave per query. Each lane owns a
// 256-point chunk of d^2 cached in LDS; 64 rounds of wave-min + owner rescan.
// ---------------------------------------------------------------------------
__global__ __launch_bounds__(32) void nbr_kernel(const float* __restrict__ pos,
                                                 const int* __restrict__ idx,
                                                 int* __restrict__ nbr) {
  __shared__ float d2buf[NPTS];
  const int q = blockIdx.x;
  const int lane = threadIdx.x;
  const int c = idx[q];
  const float cx = pos[c * 3 + 0], cy = pos[c * 3 + 1], cz = pos[c * 3 + 2];

  const int base = lane * (NPTS / 32);
  unsigned long long lkey = ~0ull;
  for (int i = 0; i < NPTS / 32; ++i) {
    const int p = base + i;
    const float dx = pos[p * 3 + 0] - cx;
    const float dy = pos[p * 3 + 1] - cy;
    const float dz = pos[p * 3 + 2] - cz;
    const float d = dx * dx + dy * dy + dz * dz;
    d2buf[p] = d;
    const unsigned long long key =
        ((unsigned long long)__float_as_uint(d) << 32) | (unsigned)p;
    lkey = (key < lkey) ? key : lkey;
  }
  __syncthreads();

  for (int k = 0; k < KNBR; ++k) {
    unsigned long long best = lkey;
#pragma unroll
    for (int off = 16; off; off >>= 1) {
      unsigned long long o = __shfl_xor(best, off, 32);
      best = (o < best) ? o : best;
    }
    const int p = (int)(unsigned)best;
    const float d = __uint_as_float((unsigned)(best >> 32));
    if (lane == 0) nbr[q * KNBR + k] = (d <= R2) ? p : -1;
    if ((p >> 8) == lane) {  // owner evicts winner, rescans its chunk only
      d2buf[p] = __uint_as_float(0x7F800000u);
      unsigned long long nk = ~0ull;
      for (int i = 0; i < NPTS / 32; ++i) {
        const int pp = base + i;
        const unsigned long long key =
            ((unsigned long long)__float_as_uint(d2buf[pp]) << 32) |
            (unsigned)pp;
        nk = (key < nk) ? key : nk;
      }
      lkey = nk;
    }
  }
}

// ---------------------------------------------------------------------------
// Kernel 3: fp32 -> f16 weight conversion (W1 zero-padded 9->32 rows so layer
// 1 is a single K=32 WMMA).
// ---------------------------------------------------------------------------
__global__ __launch_bounds__(256) void cvt_kernel(
    const float* __restrict__ W1, const float* __restrict__ W2,
    const float* __restrict__ W3, _Float16* __restrict__ W1h,
    _Float16* __restrict__ W2h, _Float16* __restrict__ W3h) {
  const int t = blockIdx.x * 256 + threadIdx.x;
  if (t < 32 * 64) {
    const int r = t >> 6, cc = t & 63;
    W1h[t] = (_Float16)((r < 9) ? W1[r * 64 + cc] : 0.f);
  }
  if (t < 64 * 64) W2h[t] = (_Float16)W2[t];
  if (t < 64 * 128) W3h[t] = (_Float16)W3[t];
}

// ---------------------------------------------------------------------------
// Kernel 4: gather + 3-layer MLP + masked max over K.
// One workgroup = TWO queries (2x64 neighbor rows), 8 waves / 256 threads:
// waves 0-3 serve query q0, waves 4-7 serve q0+1; both share one copy of the
// LDS-staged weights (async DMA, ASYNCcnt-fenced). 224 WMMAs per workgroup.
// ---------------------------------------------------------------------------
__global__ __launch_bounds__(256) void mlp_kernel(
    const float* __restrict__ x, const float* __restrict__ pos,
    const int* __restrict__ idx, const int* __restrict__ nbr,
    const _Float16* __restrict__ W1h, const _Float16* __restrict__ W2h,
    const _Float16* __restrict__ W3h, const float* __restrict__ b1,
    const float* __restrict__ b2, const float* __restrict__ b3,
    float* __restrict__ out) {
  __shared__ _Float16 H0[2][64 * 32];
  __shared__ _Float16 H1[2][64 * 64];
  __shared__ _Float16 H2[2][64 * 64];
  __shared__ _Float16 W1s[32 * 64];
  __shared__ _Float16 W2s[64 * 64];
  __shared__ _Float16 W3s[64 * 128];
  __shared__ float b1s[64], b2s[64], b3s[128];
  __shared__ float ctr[2][3];
  __shared__ int nbrS[2][64], validS[2][64];

  const int t = threadIdx.x;
  const int q0 = blockIdx.x * 2;

  // Async-DMA the f16 weights into LDS (16B per lane per issue): overlaps with
  // the scalar/index work below, no VGPR round-trip, tracked by ASYNCcnt.
  {
    async_copy16((unsigned)(size_t)W1s + t * 16, t * 16, W1h);  // 4 KB
#pragma unroll
    for (int i = 0; i < 2; ++i) {                               // 8 KB
      const int off = i * 4096 + t * 16;
      async_copy16((unsigned)(size_t)W2s + off, off, W2h);
    }
#pragma unroll
    for (int i = 0; i < 4; ++i) {                               // 16 KB
      const int off = i * 4096 + t * 16;
      async_copy16((unsigned)(size_t)W3s + off, off, W3h);
    }
  }

  if (t < 128) {
    const int qs = t >> 6, k = t & 63;
    const int j = nbr[(q0 + qs) * KNBR + k];
    validS[qs][k] = (j >= 0);
    nbrS[qs][k] = (j >= 0) ? j : 0;
    b3s[t] = b3[t];
  }
  if (t < 64) {
    b1s[t] = b1[t];
    b2s[t] = b2[t];
  }
  if (t < 6) {
    const int qs = t / 3, d = t % 3;
    ctr[qs][d] = pos[idx[q0 + qs] * 3 + d];
  }
  asm volatile("s_wait_asynccnt 0" ::: "memory");
  __syncthreads();

  // Gather H0[qs][k][0..8] = [x_j(6), pos_j - pos_i(3)], zero-padded K to 32.
  for (int e = t; e < 2 * 64 * 32; e += 256) {
    const int qs = e >> 11;
    const int r = (e >> 5) & 63;
    const int cc = e & 31;
    float v = 0.f;
    if (validS[qs][r]) {
      const int j = nbrS[qs][r];
      if (cc < 6)      v = x[j * 6 + cc];
      else if (cc < 9) v = pos[j * 3 + (cc - 6)] - ctr[qs][cc - 6];
    }
    H0[qs][(r << 5) | cc] = (_Float16)v;
  }
  __syncthreads();

  const int wave = t >> 5;
  const int qs = wave >> 2;    // which query this wave serves
  const int nt0 = wave & 3;    // N-tile within that query
  const int lane = t & 31;
  const int nl = lane & 15;
  const int mb = (lane >> 4) * 8;

  // Layer 1: [64,32] x [32,64], K=32 -> one WMMA per tile; B reused over mt.
  {
    const v16h b = load_B_f16(W1s, 64, nt0 * 16);
    const float bias = b1s[nt0 * 16 + nl];
    for (int mt = 0; mt < 4; ++mt) {
      const v16h a = load_A_f16(&H0[qs][mt * 16 * 32], 32);
      v8f c = {};
      c = wmma32(a, b, c);
#pragma unroll
      for (int j = 0; j < 8; ++j) {
        const float v = fmaxf(c[j] + bias, 0.f);
        H1[qs][(mt * 16 + mb + j) * 64 + nt0 * 16 + nl] = (_Float16)v;
      }
    }
  }
  __syncthreads();

  // Layer 2: [64,64] x [64,64], K=64 -> two chained WMMAs per tile.
  {
    const v16h bB0 = load_B_f16(W2s, 64, nt0 * 16);
    const v16h bB1 = load_B_f16(W2s + 32 * 64, 64, nt0 * 16);
    const float bias = b2s[nt0 * 16 + nl];
    for (int mt = 0; mt < 4; ++mt) {
      const v16h a0 = load_A_f16(&H1[qs][mt * 16 * 64], 64);
      const v16h a1 = load_A_f16(&H1[qs][mt * 16 * 64 + 32], 64);
      v8f c = {};
      c = wmma32(a0, bB0, c);
      c = wmma32(a1, bB1, c);
#pragma unroll
      for (int j = 0; j < 8; ++j) {
        const float v = fmaxf(c[j] + bias, 0.f);
        H2[qs][(mt * 16 + mb + j) * 64 + nt0 * 16 + nl] = (_Float16)v;
      }
    }
  }
  __syncthreads();

  // Layer 3 + masked max over K: wave handles n-tiles {nt0, nt0+4}; the max
  // over the 64 neighbor rows folds across mt in registers.
  for (int half = 0; half < 2; ++half) {
    const int nt = nt0 + half * 4;
    const v16h bB0 = load_B_f16(W3s, 128, nt * 16);
    const v16h bB1 = load_B_f16(W3s + 32 * 128, 128, nt * 16);
    float run[8];
#pragma unroll
    for (int j = 0; j < 8; ++j) run[j] = -__builtin_inff();
    for (int mt = 0; mt < 4; ++mt) {
      const v16h a0 = load_A_f16(&H2[qs][mt * 16 * 64], 64);
      const v16h a1 = load_A_f16(&H2[qs][mt * 16 * 64 + 32], 64);
      v8f c = {};
      c = wmma32(a0, bB0, c);
      c = wmma32(a1, bB1, c);
#pragma unroll
      for (int j = 0; j < 8; ++j) {
        const int k = mt * 16 + mb + j;
        if (validS[qs][k]) run[j] = fmaxf(run[j], c[j]);
      }
    }
    float m = run[0];
#pragma unroll
    for (int j = 1; j < 8; ++j) m = fmaxf(m, run[j]);
    m = fmaxf(m, __shfl_xor(m, 16, 32));
    if (lane < 16)
      out[(q0 + qs) * 128 + nt * 16 + nl] = m + b3s[nt * 16 + nl];
  }
}

// ---------------------------------------------------------------------------
// Kernel 5: pos[idx] output + zero tail (batch[idx] is all zeros).
// ---------------------------------------------------------------------------
__global__ __launch_bounds__(256) void tail_kernel(const float* __restrict__ pos,
                                                   const int* __restrict__ idx,
                                                   float* __restrict__ out,
                                                   int out_size) {
  const int t = blockIdx.x * 256 + threadIdx.x;
  const int posBase = MSMP * 128;
  if (t < MSMP * 3) {
    const int m = t / 3, d = t % 3;
    out[posBase + t] = pos[idx[m] * 3 + d];
  }
  const int zbase = posBase + MSMP * 3;
  for (int e = zbase + t; e < out_size; e += 16384) out[e] = 0.f;
}

extern "C" void kernel_launch(void* const* d_in, const int* in_sizes, int n_in,
                              void* d_out, int out_size, void* d_ws,
                              size_t ws_size, hipStream_t stream) {
  const float* x   = (const float*)d_in[0];
  const float* pos = (const float*)d_in[1];
  // d_in[2] = batch (int64, all zeros) -- unused
  const float* W1 = (const float*)d_in[3];
  const float* b1 = (const float*)d_in[4];
  const float* W2 = (const float*)d_in[5];
  const float* b2 = (const float*)d_in[6];
  const float* W3 = (const float*)d_in[7];
  const float* b3 = (const float*)d_in[8];
  float* out = (float*)d_out;

  char* ws = (char*)d_ws;
  int* idx = (int*)ws;                                   // 16 KB
  int* nbr = (int*)(ws + 16384);                         // 1 MB
  _Float16* W1h = (_Float16*)(ws + 16384 + MSMP * KNBR * 4);
  _Float16* W2h = W1h + 32 * 64;
  _Float16* W3h = W2h + 64 * 64;

  fps_kernel<<<1, 1024, 0, stream>>>(pos, idx);
  nbr_kernel<<<MSMP, 32, 0, stream>>>(pos, idx, nbr);
  cvt_kernel<<<32, 256, 0, stream>>>(W1, W2, W3, W1h, W2h, W3h);
  mlp_kernel<<<MSMP / 2, 256, 0, stream>>>(x, pos, idx, nbr, W1h, W2h, W3h, b1,
                                           b2, b3, out);
  tail_kernel<<<64, 256, 0, stream>>>(pos, idx, out, out_size);
}
